// FB_SPDDGBN_28767690948899
// MI455X (gfx1250) — matrix operand order repaired and checked by reading, not compiled, not run on hardware
//
#include <hip/hip_runtime.h>
#include <hip/hip_bf16.h>
#include <math.h>

#define NB 128
#define BANDS 9
#define CHN 22
#define TTT 1024
#define SEG 4
#define SEGLEN 256
#define QG 36
#define NMAT 4608          /* NB*QG */
#define DFEAT 4896
#define JS24 8
#define JS16 8

typedef __attribute__((ext_vector_type(16))) _Float16 v16h;
typedef __attribute__((ext_vector_type(8)))  _Float16 v8h;
typedef __attribute__((ext_vector_type(8)))  float    v8f;
typedef __attribute__((ext_vector_type(4)))  int      v4i;

#if __has_builtin(__builtin_amdgcn_global_load_async_to_lds_b128) && __has_builtin(__builtin_amdgcn_s_wait_asynccnt)
#define USE_ASYNC_LDS 1
#else
#define USE_ASYNC_LDS 0
#endif

__device__ inline v8f wmma16(v16h a, v16h b, v8f c) {
  return __builtin_amdgcn_wmma_f32_16x16x32_f16(false, a, false, b, (short)0, c, false, false);
}
__device__ inline int ka_idx(int e, int hi) { return e + (e >= 8 ? 8 : 0) + (hi ? 8 : 0); }
__device__ inline int kb_idx(int e, int hi) { return e + (e >= 8 ? 8 : 0) + (hi ? 8 : 0); }

// ---------------- wave32 Jacobi eigensolver on an LDS-resident symmetric matrix ----------
__device__ void jacobi(float* A, float* V, int n, int sweeps) {
  int lane = threadIdx.x;
  for (int e = lane; e < n * n; e += 32) V[e] = (e / n == e % n) ? 1.f : 0.f;
  __syncthreads();
  for (int s = 0; s < sweeps; ++s) {
    for (int p = 0; p < n - 1; ++p) {
      for (int q = p + 1; q < n; ++q) {
        float apq = A[p * n + q], app = A[p * n + p], aqq = A[q * n + q];
        float c = 1.f, sn = 0.f;
        if (fabsf(apq) > 1e-14f) {
          float tau = (aqq - app) / (2.f * apq);
          float t = (tau >= 0.f ? 1.f : -1.f) / (fabsf(tau) + sqrtf(1.f + tau * tau));
          c = 1.f / sqrtf(1.f + t * t);
          sn = t * c;
        }
        __syncthreads();
        if (lane < n) {
          float aip = A[lane * n + p], aiq = A[lane * n + q];
          A[lane * n + p] = c * aip - sn * aiq;
          A[lane * n + q] = sn * aip + c * aiq;
          float vip = V[lane * n + p], viq = V[lane * n + q];
          V[lane * n + p] = c * vip - sn * viq;
          V[lane * n + q] = sn * vip + c * viq;
        }
        __syncthreads();
        if (lane < n) {
          float api = A[p * n + lane], aqi = A[q * n + lane];
          A[p * n + lane] = c * api - sn * aqi;
          A[q * n + lane] = sn * api + c * aqi;
        }
        __syncthreads();
      }
    }
  }
}

__device__ inline void mm24(const float* A, const float* B, float* C) {
  for (int e = threadIdx.x; e < 576; e += 32) {
    int r = e / 24, c = e % 24;
    float s = 0.f;
    for (int k = 0; k < 24; ++k) s += A[r * 24 + k] * B[k * 24 + c];
    C[e] = s;
  }
}

// ---------------- weight renorm ----------------
__global__ void k_renorm_w1(const float* __restrict__ w1, float* __restrict__ w1r) {
  __shared__ float sh[33];
  int j = blockIdx.x, lane = threadIdx.x;
  float v = (lane < CHN) ? w1[j * CHN + lane] : 0.f;
  sh[lane] = v * v;
  __syncthreads();
  if (lane == 0) {
    float s = 0.f;
    for (int k = 0; k < CHN; ++k) s += sh[k];
    sh[32] = fminf(1.f, 1.f / fmaxf(sqrtf(s), 1e-12f));
  }
  __syncthreads();
  if (lane < CHN) w1r[j * CHN + lane] = v * sh[32];
}

__global__ void k_renorm_w2(const float* __restrict__ w2, _Float16* __restrict__ w2h) {
  __shared__ float sh[128];
  int j = blockIdx.x, tid = threadIdx.x;
  float s = 0.f;
  for (int k = tid; k < 400; k += 128) { float v = w2[(size_t)j * 400 + k]; s += v * v; }
  sh[tid] = s; __syncthreads();
  for (int t = 64; t > 0; t >>= 1) { if (tid < t) sh[tid] += sh[tid + t]; __syncthreads(); }
  float sc = fminf(1.f, 1.f / fmaxf(sqrtf(sh[0]), 1e-12f));
  for (int k = tid; k < 416; k += 128)
    w2h[(size_t)j * 416 + k] = (_Float16)(k < 400 ? w2[(size_t)j * 400 + k] * sc : 0.f);
}

// ---------------- conv1 (16x22 spatial GEMM per band), f16 output ----------------
__global__ void k_conv1(const float* __restrict__ X, const float* __restrict__ w1r,
                        const float* __restrict__ b1, _Float16* __restrict__ y1h) {
  int t = blockIdx.x * 256 + threadIdx.x;
  int j = blockIdx.y;
  int n = blockIdx.z;
  int b = j >> 4;
  float acc = b1[j];
  const float* xp = X + ((size_t)(n * BANDS + b) * CHN) * TTT + t;
  const float* wp = w1r + j * CHN;
  for (int ch = 0; ch < CHN; ++ch) acc += wp[ch] * xp[(size_t)ch * TTT];
  y1h[((size_t)n * 144 + j) * TTT + t] = (_Float16)acc;
}

// ---------------- batch-norm statistics (train mode, biased var) ----------------
__global__ void k_bn_stats_f16(const _Float16* __restrict__ y, const float* __restrict__ g,
                               const float* __restrict__ be, float* __restrict__ sshift, int C) {
  __shared__ float s1[256], s2[256];
  int j = blockIdx.x, tid = threadIdx.x;
  float a = 0.f, b = 0.f;
  for (int i = tid; i < NB * TTT; i += 256) {
    int n = i >> 10, t = i & 1023;
    float v = (float)y[((size_t)n * C + j) * TTT + t];
    a += v; b += v * v;
  }
  s1[tid] = a; s2[tid] = b; __syncthreads();
  for (int k = 128; k > 0; k >>= 1) { if (tid < k) { s1[tid] += s1[tid + k]; s2[tid] += s2[tid + k]; } __syncthreads(); }
  if (tid == 0) {
    float cnt = (float)(NB * TTT);
    float m = s1[0] / cnt;
    float var = s2[0] / cnt - m * m;
    float sc = g[j] * rsqrtf(var + 1e-5f);
    sshift[j] = sc;
    sshift[C + j] = be[j] - m * sc;
  }
}

// in-place BN application on an f16 activation tensor
__global__ void k_bn_apply(_Float16* __restrict__ y, const float* __restrict__ ss, int C) {
  size_t i = (size_t)blockIdx.x * 256 + threadIdx.x;
  int j = (int)((i >> 10) % (size_t)C);
  float v = (float)y[i] * ss[j] + ss[C + j];
  y[i] = (_Float16)v;
}

// ---------------- conv2: implicit GEMM via WMMA, LDS-staged input window ----------------
// block = 256 threads = 8 waves: 2 m-tiles x 4 t-tiles over a 64-column slab of one (n, band)
__global__ void k_conv2(const _Float16* __restrict__ y1n, const _Float16* __restrict__ w2h,
                        const float* __restrict__ b2, _Float16* __restrict__ y2h) {
  __shared__ _Float16 xn[16 * 88 + 96];   // 88-col padded window + zero pad
  __shared__ int tab[416];                // kb -> c1*88 + kt (or zero-pad base)
  int t0g = blockIdx.x * 64;
  int nb = blockIdx.y;
  int n = nb / BANDS, band = nb % BANDS;
  int tid = threadIdx.x;
  for (int kb = tid; kb < 416; kb += 256)
    tab[kb] = (kb < 400) ? (kb / 25) * 88 + (kb % 25) : (16 * 88);
  for (int i = tid; i < 96; i += 256) xn[16 * 88 + i] = (_Float16)0.f;
  const size_t ybase = ((size_t)n * 144 + band * 16) * TTT;
  for (int i = tid; i < 16 * 88; i += 256) {
    int c1 = i / 88, col = i % 88;
    int t = t0g + col - 12;
    _Float16 v = (_Float16)0.f;
    if (t >= 0 && t < TTT) v = y1n[ybase + (size_t)c1 * TTT + t];
    xn[i] = v;
  }
  __syncthreads();
  int w = tid >> 5, lane = tid & 31, lm = lane & 15, hi = (lane >> 4) & 1;
  int m0 = (w >> 2) * 16;
  int tloc = (w & 3) * 16;
  v8f acc = {};
  const size_t wrow = (size_t)(band * 32 + m0 + lm) * 416;
  for (int kc = 0; kc < 13; ++kc) {
    if (kc < 12) __builtin_prefetch(&w2h[wrow + (kc + 1) * 32], 0, 1);
    v16h a, b;
    v8h alo = *(const v8h*)(w2h + wrow + kc * 32 + hi * 8);
    v8h ahi = *(const v8h*)(w2h + wrow + kc * 32 + 16 + hi * 8);
#pragma unroll
    for (int e = 0; e < 8; ++e) { a[e] = alo[e]; a[e + 8] = ahi[e]; }
#pragma unroll
    for (int e = 0; e < 16; ++e) {
      int kb = kc * 32 + kb_idx(e, hi);
      b[e] = xn[tab[kb] + tloc + lm];
    }
    acc = wmma16(a, b, acc);
  }
#pragma unroll
  for (int r = 0; r < 8; ++r) {
    int ch = band * 32 + m0 + r + hi * 8;
    y2h[((size_t)n * 288 + ch) * TTT + t0g + tloc + lm] = (_Float16)(acc[r] + b2[ch]);
  }
}

// ---------------- segment covariance via WMMA (Z Z^T, K=256), async LDS staging ---------
__global__ void k_cov(const _Float16* __restrict__ y2h, const float* __restrict__ bn2,
                      float* __restrict__ S) {
  __shared__ __align__(16) _Float16 Z[32 * 256];
  __shared__ float sc2[32], sh2[32];
  int id = blockIdx.x;
  int n = id / QG, qg = id % QG;
  int b = qg / SEG, q = qg % SEG;
  int lane = threadIdx.x, lm = lane & 15, hi = lane >> 4;
  const _Float16* gsrc = y2h + ((size_t)n * 288 + b * 32) * TTT + q * SEGLEN;
  // stage raw 32x256 f16 tile (16 KB): 1024 chunks of 16 bytes
#if USE_ASYNC_LDS
  for (int i = 0; i < 32; ++i) {
    int chunk = i * 32 + lane;
    int r = chunk >> 5, c8 = chunk & 31;
    __builtin_amdgcn_global_load_async_to_lds_b128(
        (__attribute__((address_space(1))) v4i*)(gsrc + (size_t)r * TTT + c8 * 8),
        (__attribute__((address_space(3))) v4i*)(&Z[r * 256 + c8 * 8]), 0, 0);
  }
  __builtin_amdgcn_s_wait_asynccnt(0);
#else
  for (int i = 0; i < 32; ++i) {
    int chunk = i * 32 + lane;
    int r = chunk >> 5, c8 = chunk & 31;
    *(v8h*)(&Z[r * 256 + c8 * 8]) = *(const v8h*)(gsrc + (size_t)r * TTT + c8 * 8);
  }
#endif
  sc2[lane] = bn2[b * 32 + lane];
  sh2[lane] = bn2[288 + b * 32 + lane];
  __syncthreads();
  // normalize in place (one pass), then fragments are raw 128-bit LDS loads
  for (int i = lane; i < 32 * 256; i += 32) {
    int r = i >> 8;
    Z[i] = (_Float16)((float)Z[i] * sc2[r] + sh2[r]);
  }
  __syncthreads();
  const v8h* Zv = (const v8h*)Z;   // 32 rows x 32 chunks of 8 halfs
  int r0 = lm, r1 = 16 + lm;
  v8f a00 = {}, a01 = {}, a10 = {}, a11 = {};
  for (int kc = 0; kc < 8; ++kc) {
    v16h A0, A1, B0, B1;
    v8h x0 = Zv[r0 * 32 + kc * 4 + hi],      x1 = Zv[r0 * 32 + kc * 4 + 2 + hi];
    v8h y0 = Zv[r1 * 32 + kc * 4 + hi],      y1v = Zv[r1 * 32 + kc * 4 + 2 + hi];
    v8h u0 = Zv[r0 * 32 + kc * 4 + hi],  u1 = Zv[r0 * 32 + kc * 4 + 2 + hi];
    v8h w0 = Zv[r1 * 32 + kc * 4 + hi],  w1v = Zv[r1 * 32 + kc * 4 + 2 + hi];
#pragma unroll
    for (int e = 0; e < 8; ++e) {
      A0[e] = x0[e]; A0[e + 8] = x1[e];
      A1[e] = y0[e]; A1[e + 8] = y1v[e];
      B0[e] = u0[e]; B0[e + 8] = u1[e];
      B1[e] = w0[e]; B1[e + 8] = w1v[e];
    }
    a00 = wmma16(A0, B0, a00); a01 = wmma16(A0, B1, a01);
    a10 = wmma16(A1, B0, a10); a11 = wmma16(A1, B1, a11);
  }
  const float inv = 1.f / 255.f;
  float* So = S + (size_t)id * 1024;
#pragma unroll
  for (int r = 0; r < 8; ++r) {
    int rr = r + hi * 8;
    So[rr * 32 + lm]             = a00[r] * inv + (rr == lm ? 1e-5f : 0.f);
    So[rr * 32 + 16 + lm]        = a01[r] * inv;
    So[(16 + rr) * 32 + lm]      = a10[r] * inv;
    So[(16 + rr) * 32 + 16 + lm] = a11[r] * inv + (rr == lm ? 1e-5f : 0.f);
  }
}

// ---------------- BiMap1: S1 = W^T S W  (32 -> 24) via WMMA ----------------
__global__ void k_bimap1(const float* __restrict__ S, const float* __restrict__ Wb1,
                         float* __restrict__ S1) {
  __shared__ float Tl[32 * 32];
  int id = blockIdx.x, qg = id % QG;
  int lane = threadIdx.x, lm = lane & 15, hi = lane >> 4;
  const float* Sp = S + (size_t)id * 1024;
  const float* Wp = Wb1 + (size_t)qg * 768;
  for (int tm = 0; tm < 2; ++tm) {
    v16h a;
#pragma unroll
    for (int e = 0; e < 16; ++e) a[e] = (_Float16)Sp[(tm * 16 + lm) * 32 + ka_idx(e, hi)];
    for (int tn = 0; tn < 2; ++tn) {
      v16h bfr;
#pragma unroll
      for (int e = 0; e < 16; ++e) {
        int k = kb_idx(e, hi);
        int col = tn * 16 + lm;
        bfr[e] = (_Float16)(col < 24 ? Wp[k * 24 + col] : 0.f);
      }
      v8f acc = {};
      acc = wmma16(a, bfr, acc);
#pragma unroll
      for (int r = 0; r < 8; ++r) Tl[(tm * 16 + r + hi * 8) * 32 + tn * 16 + lm] = acc[r];
    }
  }
  __syncthreads();
  float* S1p = S1 + (size_t)id * 576;
  for (int tm = 0; tm < 2; ++tm) {
    v16h a;
#pragma unroll
    for (int e = 0; e < 16; ++e) {
      int k = ka_idx(e, hi);
      int row = tm * 16 + lm;
      a[e] = (_Float16)(row < 24 ? Wp[k * 24 + row] : 0.f);
    }
    for (int tn = 0; tn < 2; ++tn) {
      v16h bfr;
#pragma unroll
      for (int e = 0; e < 16; ++e) bfr[e] = (_Float16)Tl[kb_idx(e, hi) * 32 + tn * 16 + lm];
      v8f acc = {};
      acc = wmma16(a, bfr, acc);
#pragma unroll
      for (int r = 0; r < 8; ++r) {
        int row = tm * 16 + r + hi * 8, col = tn * 16 + lm;
        if (row < 24 && col < 24) S1p[row * 24 + col] = acc[r];
      }
    }
  }
}

// ---------------- SPD batch-norm building blocks ----------------
__global__ void k_meaninit(const float* __restrict__ S1, float* __restrict__ G) {
  int e = blockIdx.x * 256 + threadIdx.x;
  if (e >= 576) return;
  float s = 0.f;
  for (int i = 0; i < NMAT; ++i) s += S1[(size_t)i * 576 + e];
  G[e] = s / (float)NMAT;
}

__global__ void k_zero(float* p, int n) {
  int i = blockIdx.x * 256 + threadIdx.x;
  if (i < n) p[i] = 0.f;
}

__global__ void k_eigh_pre(const float* __restrict__ G, float* __restrict__ Gs, float* __restrict__ Gis) {
  __shared__ float A[576], V[576], fs[24], fi[24];
  int lane = threadIdx.x;
  for (int e = lane; e < 576; e += 32) A[e] = G[e];
  __syncthreads();
  jacobi(A, V, 24, JS24);
  if (lane < 24) {
    float w = fmaxf(A[lane * 24 + lane], 1e-10f);
    float sq = sqrtf(w);
    fs[lane] = sq; fi[lane] = 1.f / sq;
  }
  __syncthreads();
  for (int e = lane; e < 576; e += 32) {
    int r = e / 24, c = e % 24;
    float ss = 0.f, si = 0.f;
    for (int k = 0; k < 24; ++k) { float vv = V[r * 24 + k] * V[c * 24 + k]; ss += vv * fs[k]; si += vv * fi[k]; }
    Gs[e] = ss; Gis[e] = si;
  }
}

__global__ void k_center_log(const float* __restrict__ S1, const float* __restrict__ Gis,
                             float* __restrict__ Lacc) {
  __shared__ float A[576], V[576], B1[576], T1[576], fw[24];
  int id = blockIdx.x, lane = threadIdx.x;
  for (int e = lane; e < 576; e += 32) {
    int r = e / 24, c = e % 24;
    A[e] = 0.5f * (S1[(size_t)id * 576 + r * 24 + c] + S1[(size_t)id * 576 + c * 24 + r]);
    B1[e] = Gis[e];
  }
  __syncthreads();
  mm24(B1, A, T1); __syncthreads();
  mm24(T1, B1, V); __syncthreads();
  for (int e = lane; e < 576; e += 32) { int r = e / 24, c = e % 24; A[e] = 0.5f * (V[r * 24 + c] + V[c * 24 + r]); }
  __syncthreads();
  jacobi(A, V, 24, JS24);
  if (lane < 24) fw[lane] = logf(fmaxf(A[lane * 24 + lane], 1e-10f));
  __syncthreads();
  for (int e = lane; e < 576; e += 32) {
    int r = e / 24, c = e % 24;
    float s = 0.f;
    for (int k = 0; k < 24; ++k) s += V[r * 24 + k] * fw[k] * V[c * 24 + k];
    atomicAdd(&Lacc[e], s);
  }
}

__global__ void k_karcher_update(const float* __restrict__ Lacc, const float* __restrict__ Gs,
                                 float* __restrict__ G) {
  __shared__ float A[576], V[576], B1[576], T1[576], fw[24];
  int lane = threadIdx.x;
  const float invn = 1.f / (float)NMAT;
  for (int e = lane; e < 576; e += 32) { A[e] = Lacc[e] * invn; T1[e] = Gs[e]; }
  __syncthreads();
  jacobi(A, V, 24, JS24);
  if (lane < 24) fw[lane] = expf(A[lane * 24 + lane]);
  __syncthreads();
  for (int e = lane; e < 576; e += 32) {
    int r = e / 24, c = e % 24;
    float s = 0.f;
    for (int k = 0; k < 24; ++k) s += V[r * 24 + k] * fw[k] * V[c * 24 + k];
    B1[e] = s;
  }
  __syncthreads();
  mm24(T1, B1, A); __syncthreads();
  mm24(A, T1, V); __syncthreads();
  for (int e = lane; e < 576; e += 32) { int r = e / 24, c = e % 24; G[e] = 0.5f * (V[r * 24 + c] + V[c * 24 + r]); }
}

__global__ void k_final_center(const float* __restrict__ S1, const float* __restrict__ Gis,
                               float* __restrict__ eigw, float* __restrict__ eigV,
                               float* __restrict__ var_acc) {
  __shared__ float A[576], V[576], B1[576], T1[576], fw[24];
  int id = blockIdx.x, lane = threadIdx.x;
  for (int e = lane; e < 576; e += 32) {
    int r = e / 24, c = e % 24;
    A[e] = 0.5f * (S1[(size_t)id * 576 + r * 24 + c] + S1[(size_t)id * 576 + c * 24 + r]);
    B1[e] = Gis[e];
  }
  __syncthreads();
  mm24(B1, A, T1); __syncthreads();
  mm24(T1, B1, V); __syncthreads();
  for (int e = lane; e < 576; e += 32) { int r = e / 24, c = e % 24; A[e] = 0.5f * (V[r * 24 + c] + V[c * 24 + r]); }
  __syncthreads();
  jacobi(A, V, 24, JS24);
  if (lane < 24) {
    float w = A[lane * 24 + lane];
    eigw[(size_t)id * 24 + lane] = w;
    float lw = logf(fmaxf(w, 1e-10f));
    fw[lane] = lw * lw;
  }
  __syncthreads();
  for (int e = lane; e < 576; e += 32) eigV[(size_t)id * 576 + e] = V[e];
  if (lane == 0) {
    float s = 0.f;
    for (int k = 0; k < 24; ++k) s += fw[k];
    atomicAdd(var_acc, s);
  }
}

__global__ void k_pfinal(const float* __restrict__ var_acc, float* __restrict__ pp) {
  if (threadIdx.x == 0 && blockIdx.x == 0) pp[0] = rsqrtf(var_acc[0] / (float)NMAT + 1e-5f);
}

__global__ void k_eigh_sqrtB(const float* __restrict__ Bbn, float* __restrict__ Bs) {
  __shared__ float A[576], V[576], fw[24];
  int lane = threadIdx.x;
  for (int e = lane; e < 576; e += 32) { int r = e / 24, c = e % 24; A[e] = 0.5f * (Bbn[r * 24 + c] + Bbn[c * 24 + r]); }
  __syncthreads();
  jacobi(A, V, 24, JS24);
  if (lane < 24) fw[lane] = sqrtf(fmaxf(A[lane * 24 + lane], 1e-10f));
  __syncthreads();
  for (int e = lane; e < 576; e += 32) {
    int r = e / 24, c = e % 24;
    float s = 0.f;
    for (int k = 0; k < 24; ++k) s += V[r * 24 + k] * fw[k] * V[c * 24 + k];
    Bs[e] = s;
  }
}

__global__ void k_xn_reeig(const float* __restrict__ eigw, const float* __restrict__ eigV,
                           const float* __restrict__ Bs, const float* __restrict__ pp,
                           float* __restrict__ S2) {
  __shared__ float A[576], V[576], B1[576], T1[576], fw[24];
  int id = blockIdx.x, lane = threadIdx.x;
  float p = pp[0];
  for (int e = lane; e < 576; e += 32) { T1[e] = eigV[(size_t)id * 576 + e]; B1[e] = Bs[e]; }
  if (lane < 24) {
    float w = fmaxf(eigw[(size_t)id * 24 + lane], 1e-10f);
    fw[lane] = expf(p * logf(w));
  }
  __syncthreads();
  for (int e = lane; e < 576; e += 32) {
    int r = e / 24, c = e % 24;
    float s = 0.f;
    for (int k = 0; k < 24; ++k) s += T1[r * 24 + k] * fw[k] * T1[c * 24 + k];
    A[e] = s;
  }
  __syncthreads();
  mm24(B1, A, V);  __syncthreads();
  mm24(V, B1, T1); __syncthreads();
  for (int e = lane; e < 576; e += 32) { int r = e / 24, c = e % 24; A[e] = 0.5f * (T1[r * 24 + c] + T1[c * 24 + r]); }
  __syncthreads();
  jacobi(A, V, 24, JS24);
  if (lane < 24) fw[lane] = fmaxf(A[lane * 24 + lane], 1e-4f);
  __syncthreads();
  for (int e = lane; e < 576; e += 32) {
    int r = e / 24, c = e % 24;
    float s = 0.f;
    for (int k = 0; k < 24; ++k) s += V[r * 24 + k] * fw[k] * V[c * 24 + k];
    S2[(size_t)id * 576 + e] = s;
  }
}

// ---------------- BiMap2 (24 -> 16) via WMMA ----------------
__global__ void k_bimap2(const float* __restrict__ S2, const float* __restrict__ Wb2,
                         float* __restrict__ S2b) {
  __shared__ float T2l[32 * 16];
  int id = blockIdx.x, qg = id % QG;
  int lane = threadIdx.x, lm = lane & 15, hi = lane >> 4;
  const float* Sp = S2 + (size_t)id * 576;
  const float* Wp = Wb2 + (size_t)qg * 384;
  for (int tm = 0; tm < 2; ++tm) {
    v16h a, bfr;
#pragma unroll
    for (int e = 0; e < 16; ++e) {
      int k = ka_idx(e, hi);
      int row = tm * 16 + lm;
      a[e] = (_Float16)((row < 24 && k < 24) ? Sp[row * 24 + k] : 0.f);
      int kb = kb_idx(e, hi);
      bfr[e] = (_Float16)(kb < 24 ? Wp[kb * 16 + lm] : 0.f);
    }
    v8f acc = {};
    acc = wmma16(a, bfr, acc);
#pragma unroll
    for (int r = 0; r < 8; ++r) T2l[(tm * 16 + r + hi * 8) * 16 + lm] = acc[r];
  }
  __syncthreads();
  v16h a, bfr;
#pragma unroll
  for (int e = 0; e < 16; ++e) {
    int k = ka_idx(e, hi);
    a[e] = (_Float16)(k < 24 ? Wp[k * 16 + lm] : 0.f);
    bfr[e] = (_Float16)T2l[kb_idx(e, hi) * 16 + lm];
  }
  v8f acc = {};
  acc = wmma16(a, bfr, acc);
#pragma unroll
  for (int r = 0; r < 8; ++r) S2b[(size_t)id * 256 + (r + hi * 8) * 16 + lm] = acc[r];
}

// ---------------- ReEig + LogEig + upper-tri features ----------------
__global__ void k_relogup(const float* __restrict__ S2b, float* __restrict__ feat) {
  __shared__ float A[256], V[256], fw[16];
  int id = blockIdx.x;
  int n = id / QG, qg = id % QG;
  int lane = threadIdx.x;
  for (int e = lane; e < 256; e += 32) {
    int r = e / 16, c = e % 16;
    A[e] = 0.5f * (S2b[(size_t)id * 256 + r * 16 + c] + S2b[(size_t)id * 256 + c * 16 + r]);
  }
  __syncthreads();
  jacobi(A, V, 16, JS16);
  if (lane < 16) fw[lane] = logf(fmaxf(A[lane * 16 + lane], 1e-4f));
  __syncthreads();
  int u = 0;
  for (int r = 0; r < 16; ++r) {
    for (int c = r; c < 16; ++c) {
      if ((u & 31) == lane) {
        float s = 0.f;
        for (int k = 0; k < 16; ++k) s += V[r * 16 + k] * fw[k] * V[c * 16 + k];
        feat[(size_t)n * DFEAT + qg * 136 + u] = s * (r == c ? 1.f : 1.41421356237f);
      }
      ++u;
    }
  }
}

// ---------------- classifier head + softmax ----------------
__global__ void k_head(const float* __restrict__ feat, const float* __restrict__ Wout,
                       const float* __restrict__ bout, float* __restrict__ out) {
  __shared__ float sh[4][256];
  int n = blockIdx.x, tid = threadIdx.x;
  float p[4] = {0.f, 0.f, 0.f, 0.f};
  for (int d = tid; d < DFEAT; d += 256) {
    float f = feat[(size_t)n * DFEAT + d];
#pragma unroll
    for (int k = 0; k < 4; ++k) p[k] += f * Wout[(size_t)k * DFEAT + d];
  }
#pragma unroll
  for (int k = 0; k < 4; ++k) sh[k][tid] = p[k];
  __syncthreads();
  for (int s = 128; s > 0; s >>= 1) {
    if (tid < s) {
#pragma unroll
      for (int k = 0; k < 4; ++k) sh[k][tid] += sh[k][tid + s];
    }
    __syncthreads();
  }
  if (tid == 0) {
    float l[4], mx = -1e30f;
    for (int k = 0; k < 4; ++k) { l[k] = sh[k][0] + bout[k]; mx = fmaxf(mx, l[k]); }
    float se = 0.f;
    for (int k = 0; k < 4; ++k) { l[k] = expf(l[k] - mx); se += l[k]; }
    for (int k = 0; k < 4; ++k) out[n * 4 + k] = l[k] / se;
  }
}

extern "C" void kernel_launch(void* const* d_in, const int* in_sizes, int n_in,
                              void* d_out, int out_size, void* d_ws, size_t ws_size,
                              hipStream_t stream) {
  const float* X    = (const float*)d_in[0];
  const float* w1   = (const float*)d_in[1];
  const float* b1   = (const float*)d_in[2];
  const float* g1   = (const float*)d_in[3];
  const float* be1  = (const float*)d_in[4];
  const float* w2   = (const float*)d_in[5];
  const float* b2   = (const float*)d_in[6];
  const float* g2   = (const float*)d_in[7];
  const float* be2  = (const float*)d_in[8];
  const float* Wbi1 = (const float*)d_in[9];
  const float* Bbn  = (const float*)d_in[10];
  const float* Wbi2 = (const float*)d_in[11];
  const float* Wout = (const float*)d_in[12];
  const float* bout = (const float*)d_in[13];
  float* out = (float*)d_out;

  float* base = (float*)d_ws;
  size_t off = 0;
  float* w1r  = base + off; off += 144 * 22;
  float* w2hf = base + off; off += 59904;                  // 9*32*416 halves
  float* bn1  = base + off; off += 288;
  float* bn2  = base + off; off += 576;
  float* Sbuf = base + off; off += (size_t)NMAT * 1024;
  float* S1   = base + off; off += (size_t)NMAT * 576;
  float* G    = base + off; off += 576;
  float* Gs   = base + off; off += 576;
  float* Gis  = base + off; off += 576;
  float* Lacc = base + off; off += 576;
  float* Bs   = base + off; off += 576;
  float* vacc = base + off; off += 1;
  float* pp   = base + off; off += 7;
  float* regA = base + off; off += 9437184;                // y1h (f16), later tail overlay
  _Float16* y2h = (_Float16*)(base + off);                 // 128*288*1024 halves
  _Float16* w2h = (_Float16*)w2hf;
  _Float16* y1h = (_Float16*)regA;

  float* eigw = regA;                                      // tail overlay (y1h dead)
  float* eigV = regA + (size_t)NMAT * 24;
  float* S2   = eigV + (size_t)NMAT * 576;
  float* S2b  = S2 + (size_t)NMAT * 576;
  float* feat = S2b + (size_t)NMAT * 256;

  // FE block
  k_renorm_w1<<<144, 32, 0, stream>>>(w1, w1r);
  k_renorm_w2<<<288, 128, 0, stream>>>(w2, w2h);
  k_conv1<<<dim3(4, 144, NB), 256, 0, stream>>>(X, w1r, b1, y1h);
  k_bn_stats_f16<<<144, 256, 0, stream>>>(y1h, g1, be1, bn1, 144);
  k_bn_apply<<<73728, 256, 0, stream>>>(y1h, bn1, 144);
  k_conv2<<<dim3(16, NB * BANDS), 256, 0, stream>>>(y1h, w2h, b2, y2h);
  k_bn_stats_f16<<<288, 256, 0, stream>>>(y2h, g2, be2, bn2, 288);

  // E2R + BiMap1
  k_cov<<<NMAT, 32, 0, stream>>>(y2h, bn2, Sbuf);
  k_bimap1<<<NMAT, 32, 0, stream>>>(Sbuf, Wbi1, S1);

  // Riemannian batch norm: Karcher flow
  k_meaninit<<<3, 256, 0, stream>>>(S1, G);
  for (int it = 0; it < 5; ++it) {
    k_eigh_pre<<<1, 32, 0, stream>>>(G, Gs, Gis);
    k_zero<<<3, 256, 0, stream>>>(Lacc, 576);
    k_center_log<<<NMAT, 32, 0, stream>>>(S1, Gis, Lacc);
    k_karcher_update<<<1, 32, 0, stream>>>(Lacc, Gs, G);
  }
  k_eigh_pre<<<1, 32, 0, stream>>>(G, Gs, Gis);
  k_zero<<<1, 32, 0, stream>>>(vacc, 1);
  k_final_center<<<NMAT, 32, 0, stream>>>(S1, Gis, eigw, eigV, vacc);
  k_pfinal<<<1, 1, 0, stream>>>(vacc, pp);
  k_eigh_sqrtB<<<1, 32, 0, stream>>>(Bbn, Bs);
  k_xn_reeig<<<NMAT, 32, 0, stream>>>(eigw, eigV, Bs, pp, S2);

  // latent + head
  k_bimap2<<<NMAT, 32, 0, stream>>>(S2, Wbi2, S2b);
  k_relogup<<<NMAT, 32, 0, stream>>>(S2b, feat);
  k_head<<<NB, 256, 0, stream>>>(feat, Wout, bout, out);

  (void)in_sizes; (void)n_in; (void)out_size; (void)ws_size;
}
